// MultiHeadAttention_180388627186
// MI455X (gfx1250) — matrix operand order repaired and checked by reading, not compiled, not run on hardware
//
#include <hip/hip_runtime.h>
#include <hip/hip_bf16.h>

#define BATCH  4
#define T_SEQ  2048
#define DMODEL 1024
#define NH     16
#define DK     64

typedef __attribute__((ext_vector_type(16))) __bf16 v16bf;
typedef __attribute__((ext_vector_type(8)))  float  v8f;
typedef __attribute__((ext_vector_type(4)))  int    i4;

union Frag16 { unsigned u[8]; v16bf v; };

#if defined(__has_builtin)
#if __has_builtin(__builtin_amdgcn_global_load_async_to_lds_b128) && \
    __has_builtin(__builtin_amdgcn_s_wait_asynccnt)
#define HAVE_ASYNC 1
#endif
#endif
#ifndef HAVE_ASYNC
#define HAVE_ASYNC 0
#endif

typedef __attribute__((address_space(1))) i4* gptr128;   // global int4*
typedef __attribute__((address_space(3))) i4* sptr128;   // LDS int4*

__device__ __forceinline__ gptr128 as_global(const void* p) {
    return (gptr128)(unsigned long long)(size_t)p;
}
__device__ __forceinline__ sptr128 as_shared(void* p) {
    return (sptr128)(unsigned)(size_t)p;   // generic LDS addr: low 32 bits = LDS offset
}

__device__ __forceinline__ __bf16 f2bf(float f) {
    union { float f; unsigned u; } x; x.f = f;
    unsigned r = x.u + 0x7FFFu + ((x.u >> 16) & 1u);   // round-to-nearest-even
    unsigned short h = (unsigned short)(r >> 16);
    __bf16 b; __builtin_memcpy(&b, &h, 2);
    return b;
}

__device__ __forceinline__ v8f wmma_bf16(const v16bf a, const v16bf b, v8f c) {
    return __builtin_amdgcn_wmma_f32_16x16x32_bf16(
        false, a, false, b, (short)0, c, false, false);
}

// ---------------- fp32 -> bf16 convert ----------------
__global__ void cvt_bf16(const float* __restrict__ src, __bf16* __restrict__ dst, long n) {
    long i = (long)blockIdx.x * blockDim.x + threadIdx.x;
    long stride = (long)gridDim.x * blockDim.x;
    for (; i < n; i += stride) dst[i] = f2bf(src[i]);
}

// src [H_, R, C] f32  ->  dst [H_, C, R] bf16
__global__ void cvtT_bf16(const float* __restrict__ src, __bf16* __restrict__ dst,
                          int H_, int R, int C) {
    long n = (long)H_ * R * C;
    long i = (long)blockIdx.x * blockDim.x + threadIdx.x;
    long stride = (long)gridDim.x * blockDim.x;
    for (; i < n; i += stride) {
        int c = (int)(i % C);
        long t = i / C;
        int r = (int)(t % R);
        int h = (int)(t / R);
        dst[((long)h * C + c) * R + r] = f2bf(src[i]);
    }
}

// ---------------- QKV projections (register double-buffered) ----------------
__global__ __launch_bounds__(128)
void proj_kernel(const __bf16* __restrict__ Xall, const __bf16* __restrict__ Wall,
                 const float* __restrict__ bq, const float* __restrict__ bk,
                 const float* __restrict__ bv,
                 __bf16* __restrict__ qp, __bf16* __restrict__ kp,
                 __bf16* __restrict__ vpt) {
    const int p    = blockIdx.z;
    const int bh   = blockIdx.y;
    const int b    = bh / NH, h = bh % NH;
    const int wave = threadIdx.x >> 5;
    const int lane = threadIdx.x & 31;
    const int m0   = (blockIdx.x * 4 + wave) * 16;
    const int mr   = lane & 15;
    const int hi   = lane >> 4;

    const __bf16* X  = Xall + (size_t)p * BATCH * T_SEQ * DMODEL
                            + (size_t)b * T_SEQ * DMODEL;
    const __bf16* Wt = Wall + (size_t)p * NH * DK * DMODEL
                            + (size_t)h * DK * DMODEL;
    const float* bias = (p == 0 ? bq : (p == 1 ? bk : bv)) + h * DK;

    const unsigned* Ar = (const unsigned*)(X + (size_t)(m0 + mr) * DMODEL);

    v8f acc[4] = {};
    Frag16 a0, a1, b0[4], b1[4];

#define LOAD_A(FR, KK) do {                                                    \
    const unsigned* _ar = Ar + ((KK) >> 1);                                    \
    _Pragma("unroll") for (int i = 0; i < 4; ++i) (FR).u[i]   = _ar[4*hi + i]; \
    _Pragma("unroll") for (int i = 0; i < 4; ++i) (FR).u[4+i] = _ar[8 + 4*hi + i]; \
} while (0)
#define LOAD_B(FR, KK) do {                                                    \
    _Pragma("unroll") for (int j = 0; j < 4; ++j) {                            \
        const unsigned* _br = (const unsigned*)(Wt + (size_t)(j*16 + mr) * DMODEL + (KK)); \
        _Pragma("unroll") for (int i = 0; i < 8; ++i) (FR)[j].u[i] = _br[(i & 4) * 2 + 4*hi + (i & 3)]; \
    }                                                                          \
} while (0)

    LOAD_A(a0, 0); LOAD_B(b0, 0);
    for (int kk = 0; kk < DMODEL; kk += 64) {
        LOAD_A(a1, kk + 32); LOAD_B(b1, kk + 32);
#pragma unroll
        for (int j = 0; j < 4; ++j) acc[j] = wmma_bf16(a0.v, b0[j].v, acc[j]);
        if (kk + 64 < DMODEL) { LOAD_A(a0, kk + 64); LOAD_B(b0, kk + 64); }
#pragma unroll
        for (int j = 0; j < 4; ++j) acc[j] = wmma_bf16(a1.v, b1[j].v, acc[j]);
    }

#pragma unroll
    for (int j = 0; j < 4; ++j) {
        float badd = bias[j * 16 + mr];
#pragma unroll
        for (int r = 0; r < 8; ++r) {
            float val = acc[j][r] + badd;
            int m = m0 + r + 8 * hi;
            int n = j * 16 + mr;
            if (p == 2)       vpt[((size_t)bh * DK + n) * T_SEQ + m] = f2bf(val);
            else if (p == 0)  qp[((size_t)bh * T_SEQ + m) * DK + n]  = f2bf(val);
            else              kp[((size_t)bh * T_SEQ + m) * DK + n]  = f2bf(val);
        }
    }
}

// ------- async (or fallback) staging of one 32-key K/V tile into LDS --------
// buf layout: [0..2047]   K tile: 32 keys x 64 dk (row-major)
//             [2048..4095] Vt tile: 64 dv x 32 s  (row-major)
__device__ __forceinline__ void stage_kv(__bf16* buf, const __bf16* Kg,
                                         const __bf16* Vg, int s0, int tid) {
    const __bf16* ksrc = Kg + (size_t)s0 * DK;   // contiguous 4KB
#if HAVE_ASYNC
#pragma unroll
    for (int c = tid; c < 256; c += 128)
        __builtin_amdgcn_global_load_async_to_lds_b128(
            as_global(ksrc + c * 8), as_shared(buf + c * 8), 0, 0);
#pragma unroll
    for (int c = tid; c < 256; c += 128) {
        int row = c >> 2, off = (c & 3) * 8;
        __builtin_amdgcn_global_load_async_to_lds_b128(
            as_global(Vg + (size_t)row * T_SEQ + s0 + off),
            as_shared(buf + 2048 + row * 32 + off), 0, 0);
    }
#else
#pragma unroll
    for (int c = tid; c < 256; c += 128)
        *(uint4*)(buf + c * 8) = *(const uint4*)(ksrc + c * 8);
#pragma unroll
    for (int c = tid; c < 256; c += 128) {
        int row = c >> 2, off = (c & 3) * 8;
        *(uint4*)(buf + 2048 + row * 32 + off) =
            *(const uint4*)(Vg + (size_t)row * T_SEQ + s0 + off);
    }
#endif
}

// ---------------- flash attention, LDS-staged K/V, double-buffered ----------
__global__ __launch_bounds__(128)
void attn_kernel(const __bf16* __restrict__ qp, const __bf16* __restrict__ kp,
                 const __bf16* __restrict__ vpt, __bf16* __restrict__ xb) {
    __shared__ __align__(16) __bf16 kv[2][4096];     // 16 KB: K+Vt tiles, 2 stages
    __shared__ __align__(16) __bf16 pbuf[4][16 * 32];// 4 KB: per-wave P staging

    const int tid  = threadIdx.x;
    const int bh   = blockIdx.y;
    const int b    = bh / NH, h = bh % NH;
    const int wave = tid >> 5;
    const int lane = tid & 31;
    const int m0   = (blockIdx.x * 4 + wave) * 16;
    const int mr   = lane & 15;
    const int hi   = lane >> 4;
    const float scale = 0.125f;

    const __bf16* Q  = qp  + (size_t)bh * T_SEQ * DK;
    const __bf16* K  = kp  + (size_t)bh * T_SEQ * DK;
    const __bf16* Vt = vpt + (size_t)bh * DK * T_SEQ;

    Frag16 qa[2];
#pragma unroll
    for (int c = 0; c < 2; ++c) {
        const unsigned* arow = (const unsigned*)(Q + (size_t)(m0 + mr) * DK + 32 * c);
#pragma unroll
        for (int i = 0; i < 4; ++i) qa[c].u[i]     = arow[4 * hi + i];
#pragma unroll
        for (int i = 0; i < 4; ++i) qa[c].u[4 + i] = arow[8 + 4 * hi + i];
    }

    v8f acc[4] = {};
    float mrow[8], lrow[8];
#pragma unroll
    for (int r = 0; r < 8; ++r) { mrow[r] = -1e30f; lrow[r] = 0.f; }

    stage_kv(kv[0], K, Vt, 0, tid);          // prologue: tile 0 in flight

    const int nTiles = T_SEQ / 32;
    for (int it = 0; it < nTiles; ++it) {
        const int s0  = it * 32;
        const int cur = it & 1;
        if (it + 1 < nTiles) {
            stage_kv(kv[cur ^ 1], K, Vt, s0 + 32, tid);
#if HAVE_ASYNC
            __builtin_amdgcn_s_wait_asynccnt(4);   // drain tile it, keep it+1 in flight
#endif
        } else {
#if HAVE_ASYNC
            __builtin_amdgcn_s_wait_asynccnt(0);
#endif
        }
        __syncthreads();

        const unsigned* kt = (const unsigned*)kv[cur];          // 32 dwords/row
        const unsigned* vt = (const unsigned*)(kv[cur] + 2048); // 16 dwords/row

        v8f c0 = {}, c1 = {};
#pragma unroll
        for (int c = 0; c < 2; ++c) {
            Frag16 kb0, kb1;
#pragma unroll
            for (int i = 0; i < 8; ++i) kb0.u[i] = kt[mr * 32 + 16 * c + (i & 4) * 2 + 4 * hi + (i & 3)];
#pragma unroll
            for (int i = 0; i < 8; ++i) kb1.u[i] = kt[(16 + mr) * 32 + 16 * c + (i & 4) * 2 + 4 * hi + (i & 3)];
            c0 = wmma_bf16(qa[c].v, kb0.v, c0);
            c1 = wmma_bf16(qa[c].v, kb1.v, c1);
        }

        float alpha[8];
#pragma unroll
        for (int r = 0; r < 8; ++r) { c0[r] *= scale; c1[r] *= scale; }
#pragma unroll
        for (int r = 0; r < 8; ++r) {
            float mx = fmaxf(c0[r], c1[r]);
#pragma unroll
            for (int off = 1; off < 16; off <<= 1)
                mx = fmaxf(mx, __shfl_xor(mx, off, 32));
            float mn = fmaxf(mrow[r], mx);
            float al = __expf(mrow[r] - mn);
            float p0 = __expf(c0[r] - mn);
            float p1 = __expf(c1[r] - mn);
            c0[r] = p0; c1[r] = p1;
            float rs = p0 + p1;
#pragma unroll
            for (int off = 1; off < 16; off <<= 1)
                rs += __shfl_xor(rs, off, 32);
            lrow[r] = lrow[r] * al + rs;
            mrow[r] = mn;
            alpha[r] = al;
        }
#pragma unroll
        for (int j = 0; j < 4; ++j)
#pragma unroll
            for (int r = 0; r < 8; ++r) acc[j][r] *= alpha[r];

        // P (16x32) C-layout -> A-layout through per-wave LDS
        __bf16* pl = pbuf[wave];
#pragma unroll
        for (int r = 0; r < 8; ++r) {
            int row = r + 8 * hi;
            pl[row * 32 + mr]      = f2bf(c0[r]);
            pl[row * 32 + 16 + mr] = f2bf(c1[r]);
        }
        __syncthreads();
        Frag16 pa;
        const unsigned* prow = (const unsigned*)(pl + mr * 32);
#pragma unroll
        for (int i = 0; i < 4; ++i) pa.u[i]     = prow[4 * hi + i];
#pragma unroll
        for (int i = 0; i < 4; ++i) pa.u[4 + i] = prow[8 + 4 * hi + i];

#pragma unroll
        for (int j = 0; j < 4; ++j) {
            Frag16 vb;
#pragma unroll
            for (int i = 0; i < 8; ++i) vb.u[i] = vt[(j * 16 + mr) * 16 + (i & 4) * 2 + 4 * hi + (i & 3)];
            acc[j] = wmma_bf16(pa.v, vb.v, acc[j]);
        }
        __syncthreads();   // all reads of kv[cur] & pbuf done before overwrite
    }

#pragma unroll
    for (int j = 0; j < 4; ++j)
#pragma unroll
        for (int r = 0; r < 8; ++r) {
            float o = acc[j][r] / lrow[r];
            int m = m0 + r + 8 * hi;
            int n = j * 16 + mr;
            xb[((size_t)b * T_SEQ + m) * DMODEL + h * DK + n] = f2bf(o);
        }
}

// ---------------- output projection (register double-buffered) --------------
__global__ __launch_bounds__(128)
void out_gemm(const __bf16* __restrict__ xb, const __bf16* __restrict__ Wot,
              const float* __restrict__ bo, float* __restrict__ out) {
    const int wave = threadIdx.x >> 5;
    const int lane = threadIdx.x & 31;
    const int mr   = lane & 15;
    const int hi   = lane >> 4;
    const int row0 = (blockIdx.x * 4 + wave) * 16;
    const int n0   = blockIdx.y * 64;

    const unsigned* Ar = (const unsigned*)(xb + (size_t)(row0 + mr) * DMODEL);

    v8f acc[4] = {};
    Frag16 a0, a1, b0[4], b1[4];

#define OLOAD_A(FR, KK) do {                                                   \
    const unsigned* _ar = Ar + ((KK) >> 1);                                    \
    _Pragma("unroll") for (int i = 0; i < 4; ++i) (FR).u[i]   = _ar[4*hi + i]; \
    _Pragma("unroll") for (int i = 0; i < 4; ++i) (FR).u[4+i] = _ar[8 + 4*hi + i]; \
} while (0)
#define OLOAD_B(FR, KK) do {                                                   \
    _Pragma("unroll") for (int j = 0; j < 4; ++j) {                            \
        const unsigned* _br = (const unsigned*)(Wot + (size_t)(n0 + j*16 + mr) * DMODEL + (KK)); \
        _Pragma("unroll") for (int i = 0; i < 8; ++i) (FR)[j].u[i] = _br[(i & 4) * 2 + 4*hi + (i & 3)]; \
    }                                                                          \
} while (0)

    OLOAD_A(a0, 0); OLOAD_B(b0, 0);
    for (int kk = 0; kk < DMODEL; kk += 64) {
        __builtin_prefetch((const __bf16*)Ar + kk + 128, 0, 1);
        OLOAD_A(a1, kk + 32); OLOAD_B(b1, kk + 32);
#pragma unroll
        for (int j = 0; j < 4; ++j) acc[j] = wmma_bf16(a0.v, b0[j].v, acc[j]);
        if (kk + 64 < DMODEL) { OLOAD_A(a0, kk + 64); OLOAD_B(b0, kk + 64); }
#pragma unroll
        for (int j = 0; j < 4; ++j) acc[j] = wmma_bf16(a1.v, b1[j].v, acc[j]);
    }

#pragma unroll
    for (int j = 0; j < 4; ++j) {
        float badd = bo[n0 + j * 16 + mr];
#pragma unroll
        for (int r = 0; r < 8; ++r)
            out[(size_t)(row0 + r + 8 * hi) * DMODEL + n0 + j * 16 + mr] =
                acc[j][r] + badd;
    }
}

extern "C" void kernel_launch(void* const* d_in, const int* in_sizes, int n_in,
                              void* d_out, int out_size, void* d_ws, size_t ws_size,
                              hipStream_t stream) {
    const float* Q  = (const float*)d_in[0];
    const float* K  = (const float*)d_in[1];
    const float* V  = (const float*)d_in[2];
    const float* Wq = (const float*)d_in[3];
    const float* bq = (const float*)d_in[4];
    const float* Wk = (const float*)d_in[5];
    const float* bk = (const float*)d_in[6];
    const float* Wv = (const float*)d_in[7];
    const float* bv = (const float*)d_in[8];
    const float* Wo = (const float*)d_in[9];
    const float* bo = (const float*)d_in[10];

    __bf16* ws = (__bf16*)d_ws;
    size_t o = 0;
    const size_t nIn = (size_t)BATCH * T_SEQ * DMODEL;
    const size_t nW  = (size_t)NH * DK * DMODEL;
    __bf16* Qb  = ws + o; o += nIn;
    __bf16* Kb  = ws + o; o += nIn;
    __bf16* Vb  = ws + o; o += nIn;
    __bf16* Wqt = ws + o; o += nW;
    __bf16* Wkt = ws + o; o += nW;
    __bf16* Wvt = ws + o; o += nW;
    __bf16* Wot = ws + o; o += (size_t)DMODEL * DMODEL;
    __bf16* qp  = ws + o; o += nIn;
    __bf16* kp  = ws + o; o += nIn;
    __bf16* vpt = ws + o; o += nIn;
    __bf16* xbf = ws + o; o += nIn;

    cvt_bf16<<<1024, 256, 0, stream>>>(Q, Qb, (long)nIn);
    cvt_bf16<<<1024, 256, 0, stream>>>(K, Kb, (long)nIn);
    cvt_bf16<<<1024, 256, 0, stream>>>(V, Vb, (long)nIn);
    cvtT_bf16<<<512, 256, 0, stream>>>(Wq, Wqt, NH, DMODEL, DK);
    cvtT_bf16<<<512, 256, 0, stream>>>(Wk, Wkt, NH, DMODEL, DK);
    cvtT_bf16<<<512, 256, 0, stream>>>(Wv, Wvt, NH, DMODEL, DK);
    cvtT_bf16<<<512, 256, 0, stream>>>(Wo, Wot, 1, DMODEL, DMODEL);

    dim3 pg(T_SEQ / 64, BATCH * NH, 3);
    proj_kernel<<<pg, 128, 0, stream>>>(Qb, Wqt, bq, bk, bv, qp, kp, vpt);

    dim3 ag(T_SEQ / 64, BATCH * NH);
    attn_kernel<<<ag, 128, 0, stream>>>(qp, kp, vpt, xbf);

    dim3 og(BATCH * T_SEQ / 64, DMODEL / 64);
    out_gemm<<<og, 128, 0, stream>>>(xbf, Wot, bo, (float*)d_out);
}
